// PointNet2Refine_84670985273369
// MI455X (gfx1250) — matrix operand order repaired and checked by reading, not compiled, not run on hardware
//
#include <hip/hip_runtime.h>
#include <hip/hip_bf16.h>
#include <math.h>

// ---------------------------------------------------------------------------
// MI455X (gfx1250) implementation notes:
//  * wave32 only; WMMA 16x16x32 f16 (f32 accum) for all GEMM work.
//  * Operand roles: A = weights (pre-swizzled to A-fragment order, one
//    aligned 32B vector load per lane), B = activations (point-major [m][k]
//    LDS layout -> 16 contiguous halfs per lane). D tiles have out-features
//    down rows, points across columns, so each lane's 8 accumulators are 8
//    CONTIGUOUS out-features of one point -> packed relu+cvt and a single
//    ds_store_b128 per fragment. Both sides of every layer are vector LDS ops.
//  * One workgroup per grasp candidate keeps all activations in LDS
//    (132 KB dynamic of the 320 KB WGP pool); ~31 GFLOP total, WMMA-bound.
// ---------------------------------------------------------------------------

typedef __attribute__((ext_vector_type(16))) _Float16 v16h;
typedef __attribute__((ext_vector_type(8)))  _Float16 v8h;
typedef __attribute__((ext_vector_type(8)))  float    v8f;

#define NUM_PTS  20000
#define NUM_GR   512
#define BN_TOT   1024
#define NEI      192
#define GNUM     64
#define RAD2     0.09f      // 0.3^2
#define DEPTHS   0.3f
#define YLIM     0.15f      // WIDTHS/2
#define ZLIM     0.1f       // HEIGHT/2

// workspace layout (bytes); weight regions hold A-fragment-swizzled tiles
#define OFF_B1   0                               // K=32 (pad 3), N=512
#define OFF_B2   (OFF_B1 + 32*512*2)             // K=512, N=256
#define OFF_B3   (OFF_B2 + 512*256*2)            // K=256, N=256
#define OFF_B4   (OFF_B3 + 256*256*2)            // K=256, N=128
#define OFF_X    (OFF_B4 + 256*128*2)            // 65536x32 f16 (X tiles)
// total ws need: OFF_X + 1024*64*32*2 = 4,685,824 bytes

// ------------------------- WMMA fragment helpers ---------------------------
// Layouts per CDNA5 ISA 7.12.2 (wave32):
//  A (16x32 f16): lane row=l&15; elems 0..7 -> K=(l<16?0:8)+i; elems 8..15 ->
//                 K=(l<16?16:24)+i
//  B (32x16 f16): lane col=l&15; this code's A and B both use elem i -> K=(l<16?0:16)+i
//  C/D (16x16 f32): elem r -> (row r + (l<16?0:8), col l&15)

// weights pre-swizzled: tile (nt, kt) = contiguous 512-half block, lane's 16
// halfs contiguous -> one 32-byte vector load (2x global_load_b128).
__device__ __forceinline__ v16h load_wfrag(const _Float16* __restrict__ W,
                                           int numKt, int kt, int nt, int lane) {
  const _Float16* p = W + (((size_t)(nt * numKt + kt)) << 9) + (lane << 4);
  return *(const v16h*)p;
}

// activations, point-major [m][ldk]: lane col m=l&15, k contiguous.
__device__ __forceinline__ v16h load_act(const _Float16* __restrict__ H,
                                         int ldk, int m0, int k0, int lane) {
  const _Float16* p = H + (m0 + (lane & 15)) * ldk + k0 + ((lane < 16) ? 0 : 16);
  return *(const v16h*)p;   // 32B contiguous, 32B aligned -> 2x ds_load_b128
}

__device__ __forceinline__ v8f wmma16(v16h a, v16h b, v8f c) {
  return __builtin_amdgcn_wmma_f32_16x16x32_f16(false, a, false, b,
                                                (short)0, c, false, false);
}

// D fragment -> H[m][n] f16 with bias+relu; 8 contiguous out-features per
// lane -> packed converts + single 16B LDS store.
__device__ __forceinline__ void store_relu_h16(_Float16* __restrict__ H, int ldN,
                                               int m0, int n0, int lane, v8f c,
                                               const float* __restrict__ bias) {
  const int m  = m0 + (lane & 15);
  const int nb = n0 + ((lane < 16) ? 0 : 8);
  const v8f bv = *(const v8f*)(bias + nb);   // 32B aligned (nb multiple of 8)
  v8h h;
#pragma unroll
  for (int r = 0; r < 8; ++r) {
    float v = c[r] + bv[r];
    v = v > 0.0f ? v : 0.0f;
    h[r] = (_Float16)v;
  }
  *(v8h*)(H + m * ldN + nb) = h;             // 16B aligned -> ds_store_b128
}

__device__ __forceinline__ void store_relu_f32v(float* __restrict__ H, int ldN,
                                                int m0, int n0, int lane, v8f c,
                                                const float* __restrict__ bias) {
  const int m  = m0 + (lane & 15);
  const int nb = n0 + ((lane < 16) ? 0 : 8);
  const v8f bv = *(const v8f*)(bias + nb);
  v8f o;
#pragma unroll
  for (int r = 0; r < 8; ++r) {
    float v = c[r] + bv[r];
    o[r] = v > 0.0f ? v : 0.0f;
  }
  *(v8f*)(H + m * ldN + nb) = o;             // 32B aligned -> 2x ds_store_b128
}

// one 32-wide out-feature strip: Hout[0:64][strip*32 : +32] =
//   relu(W(strip rows) x Hin(64 points x K) + bias)
__device__ __forceinline__ void gemm_strip(const _Float16* __restrict__ W, int numKt,
                                           const _Float16* __restrict__ Hin, int ldk,
                                           const float* __restrict__ bias,
                                           _Float16* __restrict__ Hout, int ldN,
                                           int strip, int lane) {
  v8f acc[2][4];
#pragma unroll
  for (int a = 0; a < 2; ++a)
#pragma unroll
    for (int mt = 0; mt < 4; ++mt)
      acc[a][mt] = (v8f){0.f,0.f,0.f,0.f,0.f,0.f,0.f,0.f};

  const int nt0 = strip * 2;
  for (int kt = 0; kt < numKt; ++kt) {
    v16h af0 = load_wfrag(W, numKt, kt, nt0,     lane);
    v16h af1 = load_wfrag(W, numKt, kt, nt0 + 1, lane);
#pragma unroll
    for (int mt = 0; mt < 4; ++mt) {
      v16h bf = load_act(Hin, ldk, mt * 16, kt * 32, lane);
      acc[0][mt] = wmma16(af0, bf, acc[0][mt]);
      acc[1][mt] = wmma16(af1, bf, acc[1][mt]);
    }
  }
#pragma unroll
  for (int a = 0; a < 2; ++a)
#pragma unroll
    for (int mt = 0; mt < 4; ++mt)
      store_relu_h16(Hout, ldN, mt * 16, strip * 32 + a * 16, lane, acc[a][mt], bias);
}

// ------------------------- kernel 0: weight prep ---------------------------
// Write each weight matrix as WMMA-A fragment tiles:
//   dst[(nt*numKt + kt)*512 + lane*16 + e] = w[n][k]
//   n = nt*16 + (lane&15)
//   k = kt*32 + (lane<16?0:16) + e     (0 beyond Ksrc)
__device__ __forceinline__ void swz_region(_Float16* __restrict__ dst,
                                           const float* __restrict__ src,
                                           int j, int Kpad, int Ksrc) {
  const int numKt = Kpad >> 5;
  const int tile = j >> 9;
  const int lane = (j >> 4) & 31;
  const int e    = j & 15;
  const int nt = tile / numKt;
  const int kt = tile - nt * numKt;
  const int n  = nt * 16 + (lane & 15);
  const int kb = (lane < 16) ? 0 : 16;
  const int k  = kt * 32 + kb + e;
  dst[j] = (k < Ksrc) ? (_Float16)src[n * Ksrc + k] : (_Float16)0.0f;
}

__global__ void k_prep_weights(const float* __restrict__ w1, const float* __restrict__ w2,
                               const float* __restrict__ w3, const float* __restrict__ w4,
                               char* __restrict__ ws) {
  int i = blockIdx.x * blockDim.x + threadIdx.x;
  const int n1 = 32 * 512, n2 = n1 + 512 * 256, n3 = n2 + 256 * 256, n4 = n3 + 256 * 128;
  if (i < n1) {
    swz_region((_Float16*)(ws + OFF_B1), w1, i, 32, 3);
  } else if (i < n2) {
    swz_region((_Float16*)(ws + OFF_B2), w2, i - n1, 512, 512);
  } else if (i < n3) {
    swz_region((_Float16*)(ws + OFF_B3), w3, i - n2, 256, 256);
  } else if (i < n4) {
    swz_region((_Float16*)(ws + OFF_B4), w4, i - n3, 256, 256);
  }
}

// --------------- kernel 1: ball query + gripper transform ------------------
// one wave32 per candidate; ordered first-K compaction via ballot prefix.
__global__ void k_select_transform(const float* __restrict__ grasp,
                                   const float* __restrict__ pc,
                                   char* __restrict__ ws,
                                   float* __restrict__ out_mask) {
  __shared__ int   s_idx[NEI];
  __shared__ float s_px[NEI], s_py[NEI], s_pz[NEI];
  __shared__ int   s_in[NEI];
  __shared__ int   s_sel[GNUM];

  const int bn = blockIdx.x;
  const int b  = bn / NUM_GR;
  const int lane = threadIdx.x;

  const float* g = grasp + bn * 8;
  float gs = 0.f;
#pragma unroll
  for (int i = 0; i < 8; ++i) gs += g[i];
  const bool grasp_ok = (gs != -8.0f);

  const float cx = g[0], cy = g[1], cz = g[2];
  const float ang = g[6];
  const float cA = cosf(ang), sA = sinf(ang);
  float y0 = g[3], y1 = g[4], y2 = g[5];
  const float ny = sqrtf(y0*y0 + y1*y1 + y2*y2) + 1e-12f;
  y0 /= ny; y1 /= ny; y2 /= ny;
  float x0 = y1, x1 = -y0, x2 = 0.0f;
  const float nx = sqrtf(x0*x0 + x1*x1) + 1e-12f;
  x0 /= nx; x1 /= nx;
  float z0 = x1*y2 - x2*y1, z1 = x2*y0 - x0*y2, z2 = x0*y1 - x1*y0;
  const float nz = sqrtf(z0*z0 + z1*z1 + z2*z2);
  if (nz == 0.0f) { z0 = 0.f; z1 = 0.f; z2 = 1.f; }
  else           { z0 /= nz; z1 /= nz; z2 /= nz; }
  float ap0 = cA*x0 + sA*z0, ap1 = cA*x1 + sA*z1, ap2 = cA*x2 + sA*z2;
  const float na = sqrtf(ap0*ap0 + ap1*ap1 + ap2*ap2) + 1e-12f;
  ap0 /= na; ap1 /= na; ap2 /= na;
  const float mi0 = ap1*y2 - ap2*y1;
  const float mi1 = ap2*y0 - ap0*y2;
  const float mi2 = ap0*y1 - ap1*y0;

  const float* pcb = pc + (size_t)b * NUM_PTS * 6;

  // pass 1: first 192 points within radius, index order (20000 = 625*32)
  int count = 0;
  for (int a0 = 0; a0 < NUM_PTS && count < NEI; a0 += 32) {
    const int a = a0 + lane;
    const float* p = pcb + a * 6;
    const float dx = p[0] - cx, dy = p[1] - cy, dz = p[2] - cz;
    const bool v = (dx*dx + dy*dy + dz*dz) < RAD2;
    const unsigned m32 = (unsigned)__ballot(v);
    const int pre = __popc(m32 & ((1u << lane) - 1u));
    if (v) { const int pos = count + pre; if (pos < NEI) s_idx[pos] = a; }
    count += __popc(m32);
  }
  if (count > NEI) count = NEI;
  __syncthreads();
  {
    const int first = s_idx[0];
    const int pad = (count > 0) ? first : 0;
    for (int s = count + lane; s < NEI; s += 32) s_idx[s] = pad;
  }
  __syncthreads();

  // rotate all 192 (incl. padded duplicates) into gripper frame + box test
  for (int s = lane; s < NEI; s += 32) {
    const float* p = pcb + s_idx[s] * 6;
    const float vx = p[0] - cx, vy = p[1] - cy, vz = p[2] - cz;
    const float tx = ap0*vx + ap1*vy + ap2*vz;
    const float ty = y0 *vx + y1 *vy + y2 *vz;
    const float tz = mi0*vx + mi1*vy + mi2*vz;
    s_px[s] = tx; s_py[s] = ty; s_pz[s] = tz;
    s_in[s] = (tx > 0.f) && (tx < DEPTHS) && (ty > -YLIM) && (ty < YLIM) &&
              (tz > -ZLIM) && (tz < ZLIM);
  }
  __syncthreads();

  // pass 2: first 64 inside the box, slot order
  int ccnt = 0;
  for (int s0 = 0; s0 < NEI && ccnt < GNUM; s0 += 32) {
    const bool v = s_in[s0 + lane] != 0;
    const unsigned m32 = (unsigned)__ballot(v);
    const int pre = __popc(m32 & ((1u << lane) - 1u));
    if (v) { const int pos = ccnt + pre; if (pos < GNUM) s_sel[pos] = s0 + lane; }
    ccnt += __popc(m32);
  }
  if (ccnt > GNUM) ccnt = GNUM;
  __syncthreads();
  {
    const int first = s_sel[0];
    const int pad = (ccnt > 0) ? first : 0;
    for (int s = ccnt + lane; s < GNUM; s += 32) s_sel[s] = pad;
  }
  __syncthreads();

  // emit 64x32 f16 X tile (cols 0..2 = xyz, 3..31 = 0 -> K padded for WMMA)
  _Float16* X = (_Float16*)(ws + OFF_X) + (size_t)bn * GNUM * 32;
  for (int j = lane; j < GNUM; j += 32) {
    const int s = s_sel[j];
    v8h row0 = (v8h){(_Float16)s_px[s], (_Float16)s_py[s], (_Float16)s_pz[s],
                     (_Float16)0.f, (_Float16)0.f, (_Float16)0.f,
                     (_Float16)0.f, (_Float16)0.f};
    v8h zero = (v8h){(_Float16)0.f,(_Float16)0.f,(_Float16)0.f,(_Float16)0.f,
                     (_Float16)0.f,(_Float16)0.f,(_Float16)0.f,(_Float16)0.f};
    v8h* row = (v8h*)(X + j * 32);
    row[0] = row0; row[1] = zero; row[2] = zero; row[3] = zero;
  }
  if (lane == 0)
    out_mask[bn] = (grasp_ok && count > 0 && ccnt > 0) ? 1.0f : 0.0f;
}

// ------------- kernel 2: fused WMMA MLP + factored heads -------------------
__device__ __forceinline__ void head_out(float* __restrict__ dst,
                                         const float* __restrict__ w, float wb,
                                         const float* __restrict__ dcw, float dcb,
                                         const float* __restrict__ sS1,
                                         const float* __restrict__ sH4, int lane) {
  // t1 = w[0:256] . s1      (n-independent part of the head)
  float t1 = 0.f;
  for (int j = lane; j < 256; j += 32) t1 += w[j] * sS1[j];
  // hsum = sum_n dcw[n] * (w[256:384] . h4[n])   lane owns j = lane*4..+4
  float hsum = 0.f;
  const float* wh = w + 256;
  for (int n = 0; n < GNUM; ++n) {
    float d = 0.f;
#pragma unroll
    for (int q = 0; q < 4; ++q) {
      const int j = lane * 4 + q;
      d += wh[j] * sH4[n * 128 + j];
    }
    hsum += dcw[n] * d;
  }
  // ssum = sum_n dcw[n]
  float ssum = 0.f;
  for (int n = lane; n < GNUM; n += 32) ssum += dcw[n];
#pragma unroll
  for (int o = 16; o > 0; o >>= 1) {
    t1   += __shfl_xor(t1,   o);
    hsum += __shfl_xor(hsum, o);
    ssum += __shfl_xor(ssum, o);
  }
  if (lane == 0) *dst = (t1 + wb) * ssum + hsum + dcb;
}

__global__ __launch_bounds__(256) void k_mlp_heads(
    const char* __restrict__ ws, const float* __restrict__ stage1,
    const float* __restrict__ b1, const float* __restrict__ b2,
    const float* __restrict__ b3, const float* __restrict__ b4,
    const float* __restrict__ reg_w, const float* __restrict__ reg_b,
    const float* __restrict__ cls_w, const float* __restrict__ cls_b,
    const float* __restrict__ dc_reg_w, const float* __restrict__ dc_reg_b,
    const float* __restrict__ dc_cls_w, const float* __restrict__ dc_cls_b,
    float* __restrict__ out) {
  extern __shared__ char smem[];
  _Float16* sX  = (_Float16*)(smem);                           // 64x32   4 KB
  _Float16* sH1 = (_Float16*)(smem + 4096);                    // 64x512 64 KB
  _Float16* sH2 = (_Float16*)(smem + 4096 + 65536);            // 64x256 32 KB
  _Float16* sH3 = (_Float16*)(smem + 4096 + 65536 + 32768);    // 64x256 32 KB
  float*    sH4 = (float*)(smem + 4096);                       // 64x128 f32 (over dead sH1)
  float*    sS1 = (float*)(smem);                              // 256 f32 (over dead sX)

  const int bn   = blockIdx.x;
  const int tid  = threadIdx.x;
  const int wid  = tid >> 5;
  const int lane = tid & 31;

  const _Float16* B1 = (const _Float16*)(ws + OFF_B1);
  const _Float16* B2 = (const _Float16*)(ws + OFF_B2);
  const _Float16* B3 = (const _Float16*)(ws + OFF_B3);
  const _Float16* B4 = (const _Float16*)(ws + OFF_B4);
  const _Float16* X  = (const _Float16*)(ws + OFF_X) + (size_t)bn * GNUM * 32;

  // stage X tile (4 KB) with vector copies
  {
    const uint4* src = (const uint4*)X;
    uint4* dst = (uint4*)sX;
    for (int i = tid; i < (GNUM * 32 * 2) / 16; i += 256) dst[i] = src[i];
  }
  __syncthreads();

  // layer 1: H1(64x512) = relu(W1 x X)   16 strips, 2 per wave, K=32
  gemm_strip(B1, 1, sX, 32, b1, sH1, 512, wid,     lane);
  gemm_strip(B1, 1, sX, 32, b1, sH1, 512, wid + 8, lane);
  __syncthreads();

  // layer 2: H2(64x256) = relu(W2 x H1), K=512
  gemm_strip(B2, 16, sH1, 512, b2, sH2, 256, wid, lane);
  __syncthreads();

  // layer 3: H3(64x256) = relu(W3 x H2), K=256
  gemm_strip(B3, 8, sH2, 256, b3, sH3, 256, wid, lane);
  __syncthreads();

  // layer 4: H4(64x128) = relu(W4 x H3), K=256, f32 out; 1 A-tile per wave
  {
    v8f acc[4];
#pragma unroll
    for (int mt = 0; mt < 4; ++mt) acc[mt] = (v8f){0.f,0.f,0.f,0.f,0.f,0.f,0.f,0.f};
    for (int kt = 0; kt < 8; ++kt) {
      v16h af = load_wfrag(B4, 8, kt, wid, lane);
#pragma unroll
      for (int mt = 0; mt < 4; ++mt) {
        v16h bf = load_act(sH3, 256, mt * 16, kt * 32, lane);
        acc[mt] = wmma16(af, bf, acc[mt]);
      }
    }
#pragma unroll
    for (int mt = 0; mt < 4; ++mt)
      store_relu_f32v(sH4, 128, mt * 16, wid * 16, lane, acc[mt], b4);
  }
  __syncthreads();

  // stage s1 feature (256 f32) and run factored heads
  sS1[tid] = stage1[(size_t)bn * 256 + tid];
  __syncthreads();

  // reg head c = wid (8 waves -> 8 channels)
  head_out(out + 2048 + bn * 8 + wid, reg_w + wid * 384, reg_b[wid],
           dc_reg_w + wid * 64, dc_reg_b[wid], sS1, sH4, lane);
  // cls head c = wid for waves 0,1
  if (wid < 2)
    head_out(out + bn * 2 + wid, cls_w + wid * 384, cls_b[wid],
             dc_cls_w + wid * 64, dc_cls_b[wid], sS1, sH4, lane);
}

// ---------------------------------------------------------------------------
extern "C" void kernel_launch(void* const* d_in, const int* in_sizes, int n_in,
                              void* d_out, int out_size, void* d_ws, size_t ws_size,
                              hipStream_t stream) {
  (void)in_sizes; (void)n_in; (void)out_size; (void)ws_size;
  const float* grasp    = (const float*)d_in[0];
  const float* pc       = (const float*)d_in[1];
  // d_in[2] all_feature: grouped then discarded by the reference -> skipped
  const float* stage1   = (const float*)d_in[3];
  const float* w1 = (const float*)d_in[4];  const float* b1 = (const float*)d_in[5];
  const float* w2 = (const float*)d_in[6];  const float* b2 = (const float*)d_in[7];
  const float* w3 = (const float*)d_in[8];  const float* b3 = (const float*)d_in[9];
  const float* w4 = (const float*)d_in[10]; const float* b4 = (const float*)d_in[11];
  const float* reg_w = (const float*)d_in[12]; const float* reg_b = (const float*)d_in[13];
  const float* cls_w = (const float*)d_in[14]; const float* cls_b = (const float*)d_in[15];
  const float* dc_reg_w = (const float*)d_in[16]; const float* dc_reg_b = (const float*)d_in[17];
  const float* dc_cls_w = (const float*)d_in[18]; const float* dc_cls_b = (const float*)d_in[19];

  char*  ws  = (char*)d_ws;
  float* out = (float*)d_out;   // [cls 2048 | reg 8192 | mask 1024]

  const int prep_total = 32*512 + 512*256 + 256*256 + 256*128;  // 245760
  k_prep_weights<<<(prep_total + 255) / 256, 256, 0, stream>>>(w1, w2, w3, w4, ws);

  k_select_transform<<<BN_TOT, 32, 0, stream>>>(grasp, pc, ws, out + 2048 + 8192);

  // dynamic LDS: 4KB X + 64KB H1 + 32KB H2 + 32KB H3 = 132 KB (H4/s1 reuse)
  k_mlp_heads<<<BN_TOT, 256, 135168, stream>>>(
      ws, stage1, b1, b2, b3, b4, reg_w, reg_b, cls_w, cls_b,
      dc_reg_w, dc_reg_b, dc_cls_w, dc_cls_b, out);
}
